// GAT_30640296689759
// MI455X (gfx1250) — matrix operand order repaired and checked by reading, not compiled, not run on hardware
//
#include <hip/hip_runtime.h>

// ---------------- problem constants ----------------
#define Nn   20000
#define Ee   320000
#define Gg   64
#define Hh   4
#define Cc   64
#define HC   256
#define DIN  128
#define HID  256
#define LAT  64
#define ETOT (Ee + Nn)          // edges + self loops
#define EPSV 1e-5f

typedef __attribute__((ext_vector_type(16))) _Float16 v16h;
typedef __attribute__((ext_vector_type(8)))  _Float16 v8h;
typedef __attribute__((ext_vector_type(8)))  float    v8f;

// ---------------- CDNA5 async global->LDS copy (ASYNCcnt path) ----------------
__device__ __forceinline__ void async_copy_b128(const _Float16* g, _Float16* l) {
  unsigned lds = (unsigned)(unsigned long long)(void*)l;   // low 32 bits = LDS offset
  asm volatile("global_load_async_to_lds_b128 %0, %1, off"
               :: "v"(lds), "v"(g)
               : "memory");
}

// ---------------- f32 -> f16 conversions (once per launch) ----------------
__global__ void cvt_f32_f16(const float* __restrict__ in, _Float16* __restrict__ out, int n) {
  int i = blockIdx.x * blockDim.x + threadIdx.x;
  if (i < n) out[i] = (_Float16)in[i];
}

// Wt[n*K + k] = (f16) W[k*256 + n]   (transpose so LDS B-stage is contiguous)
__global__ void cvt_weight_t(const float* __restrict__ W, _Float16* __restrict__ Wt, int K) {
  int i = blockIdx.x * blockDim.x + threadIdx.x;   // over K*256
  if (i >= K * 256) return;
  int k = i >> 8, n = i & 255;
  Wt[(size_t)n * K + k] = (_Float16)W[i];
}

// ---------------- WMMA GEMM: C[M x 256] = A[M x K] * B[K x 256] ----------------
// f16 A (row-major [M,K]) and f16 Bt (row-major [256,K]) in global.
// Double-buffered LDS tiles filled by global_load_async_to_lds_b128,
// v_wmma_f32_16x16x32_f16 with f32 accumulate.
// block = 256 threads = 8 wave32; block tile 32(M) x 256(N); wave tile 16 x 64.
__launch_bounds__(256)
__global__ void gemm_wmma_f16(const _Float16* __restrict__ A,
                              const _Float16* __restrict__ Bt,
                              float* __restrict__ Cout, int K) {
  __shared__ _Float16 As[2][32 * 40];    // [row][k], stride 40 halves (80B, 16B-aligned)
  __shared__ _Float16 Bs[2][256 * 40];   // [n][k]

  const int t    = threadIdx.x;
  const int lane = t & 31;
  const int wave = t >> 5;
  const int m0   = blockIdx.x * 32;
  const int m_sub = (wave >> 2) * 16;       // 0 or 16
  const int n_sub = (wave & 3) * 64;        // 0,64,128,192

  v8f acc[4];
#pragma unroll
  for (int j = 0; j < 4; ++j)
#pragma unroll
    for (int i = 0; i < 8; ++i) acc[j][i] = 0.0f;

  const int ntiles = K >> 5;

  // exactly 5 async b128 instructions per wave per tile:
  //   A tile 32x32 halves = 128 chunks -> even lanes, 1 instr/wave
  //   B tile 256x32 halves = 1024 chunks -> 4 instr/wave
  auto stage = [&](int tile, int buf) {
    const int k0 = tile << 5;
    if ((t & 1) == 0) {
      int chunk = t >> 1;                     // 0..127
      int row = chunk >> 2, seg = chunk & 3;
      async_copy_b128(A + (size_t)(m0 + row) * K + k0 + seg * 8,
                      &As[buf][row * 40 + seg * 8]);
    }
#pragma unroll
    for (int i = 0; i < 4; ++i) {
      int chunk = t + i * 256;                // 0..1023
      int n = chunk >> 2, seg = chunk & 3;
      async_copy_b128(Bt + (size_t)n * K + k0 + seg * 8,
                      &Bs[buf][n * 40 + seg * 8]);
    }
  };

  stage(0, 0);
  for (int i = 0; i < ntiles; ++i) {
    const int cur = i & 1;
    __syncthreads();                          // prior compute on buf cur^1 finished
    if (i + 1 < ntiles) {
      stage(i + 1, cur ^ 1);                  // overlap next-tile copy with this tile's WMMA
      asm volatile("s_wait_asynccnt 0x5" ::: "memory");   // tile i done; tile i+1 in flight
    } else {
      asm volatile("s_wait_asynccnt 0x0" ::: "memory");
    }
    __syncthreads();

    // A fragment: lane l -> row m_sub+(l&15); K elems base..base+7 and 16+base..16+base+7
    const int arow  = m_sub + (lane & 15);
    const int abase = (lane >> 4) * 8;
    v8h alo = *(const v8h*)&As[cur][arow * 40 + abase];
    v8h ahi = *(const v8h*)&As[cur][arow * 40 + 16 + abase];
    v16h afrag;
#pragma unroll
    for (int q = 0; q < 8; ++q) { afrag[q] = alo[q]; afrag[q + 8] = ahi[q]; }

#pragma unroll
    for (int j = 0; j < 4; ++j) {
      const int bn  = n_sub + j * 16 + (lane & 15);
      const int klo = (lane >> 4) * 8;
      v8h blo = *(const v8h*)&Bs[cur][bn * 40 + klo];
      v8h bhi = *(const v8h*)&Bs[cur][bn * 40 + klo + 16];
      v16h bfrag;
#pragma unroll
      for (int q = 0; q < 8; ++q) { bfrag[q] = blo[q]; bfrag[q + 8] = bhi[q]; }
      acc[j] = __builtin_amdgcn_wmma_f32_16x16x32_f16(
          false, afrag, false, bfrag, (short)0, acc[j], false, false);
    }
  }

  // D layout: VGPR r -> (M = r + (lane>=16 ? 8 : 0), N = lane&15)
  const int rbase = m0 + m_sub + ((lane >> 4) * 8);
  const int cbase = n_sub + (lane & 15);
#pragma unroll
  for (int j = 0; j < 4; ++j) {
    const int col = cbase + j * 16;
#pragma unroll
    for (int r = 0; r < 8; ++r)
      Cout[(size_t)(rbase + r) * 256 + col] = acc[j][r];
  }
}

// ---------------- attention logits: al_s/al_d [N,H] ----------------
__global__ void attn_logits(const float* __restrict__ h,
                            const float* __restrict__ a_src,
                            const float* __restrict__ a_dst,
                            float* __restrict__ als, float* __restrict__ ald) {
  int idx = blockIdx.x * blockDim.x + threadIdx.x;   // n*H + head
  if (idx >= Nn * Hh) return;
  int hd = idx & 3;
  int n  = idx >> 2;
  const float* row = h + (size_t)n * HC + hd * Cc;
  const float* as_ = a_src + hd * Cc;
  const float* ad_ = a_dst + hd * Cc;
  float ss = 0.f, sd = 0.f;
#pragma unroll 8
  for (int c = 0; c < Cc; ++c) { float v = row[c]; ss += v * as_[c]; sd += v * ad_[c]; }
  als[idx] = ss;
  ald[idx] = sd;
}

// order-preserving float<->uint so atomicMax(u32) == float max; memset(0) == -inf
__device__ __forceinline__ unsigned f2ord(float f) {
  unsigned b = __float_as_uint(f);
  return (b & 0x80000000u) ? ~b : (b | 0x80000000u);
}
__device__ __forceinline__ float ord2f(unsigned u) {
  unsigned b = (u & 0x80000000u) ? (u & 0x7FFFFFFFu) : ~u;
  return __uint_as_float(b);
}

__global__ void edge_max(const int* __restrict__ src, const int* __restrict__ dst,
                         const float* __restrict__ als, const float* __restrict__ ald,
                         unsigned* __restrict__ mbuf) {
  int e = blockIdx.x * blockDim.x + threadIdx.x;
  if (e >= ETOT) return;
  int s = (e < Ee) ? src[e] : (e - Ee);
  int d = (e < Ee) ? dst[e] : (e - Ee);
#pragma unroll
  for (int hd = 0; hd < Hh; ++hd) {
    float v = als[s * Hh + hd] + ald[d * Hh + hd];
    v = v > 0.f ? v : 0.2f * v;                       // leaky_relu 0.2
    atomicMax(&mbuf[d * Hh + hd], f2ord(v));
  }
}

__global__ void edge_expsum(const int* __restrict__ src, const int* __restrict__ dst,
                            const float* __restrict__ als, const float* __restrict__ ald,
                            const unsigned* __restrict__ mbuf,
                            float* __restrict__ wbuf, float* __restrict__ sbuf) {
  int e = blockIdx.x * blockDim.x + threadIdx.x;
  if (e >= ETOT) return;
  int s = (e < Ee) ? src[e] : (e - Ee);
  int d = (e < Ee) ? dst[e] : (e - Ee);
#pragma unroll
  for (int hd = 0; hd < Hh; ++hd) {
    float v = als[s * Hh + hd] + ald[d * Hh + hd];
    v = v > 0.f ? v : 0.2f * v;
    float w = __expf(v - ord2f(mbuf[d * Hh + hd]));
    wbuf[(size_t)e * Hh + hd] = w;
    atomicAdd(&sbuf[d * Hh + hd], w);
  }
}

// one wave per edge; lane = 8 channels of one head (head width 64 = 8 lanes)
__launch_bounds__(256)
__global__ void edge_agg(const int* __restrict__ src, const int* __restrict__ dst,
                         const float* __restrict__ wbuf, const float* __restrict__ sbuf,
                         const float* __restrict__ hlin, float* __restrict__ hout) {
  int lane = threadIdx.x & 31;
  int e = blockIdx.x * 8 + (threadIdx.x >> 5);
  if (e >= ETOT) return;
  int s = (e < Ee) ? src[e] : (e - Ee);
  int d = (e < Ee) ? dst[e] : (e - Ee);
  int hd = lane >> 3;
  float alpha = wbuf[(size_t)e * Hh + hd] / (sbuf[d * Hh + hd] + 1e-16f);
  int c0 = lane * 8;
  const float4* hp = (const float4*)(hlin + (size_t)s * HC + c0);
  float4 v0 = hp[0], v1 = hp[1];
  float* op = hout + (size_t)d * HC + c0;
  atomicAdd(op + 0, alpha * v0.x); atomicAdd(op + 1, alpha * v0.y);
  atomicAdd(op + 2, alpha * v0.z); atomicAdd(op + 3, alpha * v0.w);
  atomicAdd(op + 4, alpha * v1.x); atomicAdd(op + 5, alpha * v1.y);
  atomicAdd(op + 6, alpha * v1.z); atomicAdd(op + 7, alpha * v1.w);
}

// ---------------- batchnorm over N rows x 256 cols ----------------
#define BN_ROWS 125   // 160 blocks * 125 = 20000
__global__ void bn_stats(const float* __restrict__ h, const float* __restrict__ bias,
                         float* __restrict__ sum, float* __restrict__ sumsq) {
  int col  = threadIdx.x;                 // 256 threads
  int row0 = blockIdx.x * BN_ROWS;
  float b = bias[col];
  float s = 0.f, q = 0.f;
  for (int r = 0; r < BN_ROWS; ++r) {
    float v = h[(size_t)(row0 + r) * 256 + col] + b;
    s += v; q += v * v;
  }
  atomicAdd(&sum[col], s);
  atomicAdd(&sumsq[col], q);
}

// in-place BN+ReLU on f32, plus f16 mirror for the next layer's WMMA GEMM
__global__ void bn_apply_relu(float* __restrict__ h, _Float16* __restrict__ hh,
                              const float* __restrict__ bias,
                              const float* __restrict__ sum, const float* __restrict__ sumsq,
                              const float* __restrict__ g, const float* __restrict__ beta) {
  int idx = blockIdx.x * blockDim.x + threadIdx.x;
  if (idx >= Nn * 256) return;
  int col = idx & 255;
  float mean = sum[col] * (1.0f / Nn);
  float var  = sumsq[col] * (1.0f / Nn) - mean * mean;  // biased var
  float v = (h[idx] + bias[col] - mean) * rsqrtf(var + EPSV);
  v = g[col] * v + beta[col];
  v = v > 0.f ? v : 0.f;
  h[idx]  = v;
  hh[idx] = (_Float16)v;
}

// ---------------- global add pool ----------------
__global__ void pool_sum(const float* __restrict__ h, const int* __restrict__ batch,
                         float* __restrict__ pooled) {
  int idx = blockIdx.x * blockDim.x + threadIdx.x;
  if (idx >= Nn * 256) return;
  int n = idx >> 8, col = idx & 255;
  atomicAdd(&pooled[(size_t)batch[n] * 256 + col], h[idx]);
}

// ---------------- MLP head (tiny: 64-row GEMMs) ----------------
__global__ void head_gemm_lrelu(const float* __restrict__ in, const float* __restrict__ W,
                                const float* __restrict__ bias, float* __restrict__ out,
                                int Kin, int Nout) {
  int idx = blockIdx.x * blockDim.x + threadIdx.x;
  if (idx >= Gg * Nout) return;
  int g = idx / Nout, c = idx % Nout;
  const float* ip = in + (size_t)g * Kin;
  float acc = bias[c];
  for (int k = 0; k < Kin; ++k) acc += ip[k] * W[(size_t)k * Nout + c];
  out[idx] = acc > 0.f ? acc : 0.2f * acc;
}

__global__ void head_bn(float* __restrict__ buf, const float* __restrict__ g,
                        const float* __restrict__ beta) {
  int c = threadIdx.x;                    // one block, 256 threads
  float s = 0.f, q = 0.f;
  for (int r = 0; r < Gg; ++r) { float v = buf[r * 256 + c]; s += v; q += v * v; }
  float mean = s * (1.0f / Gg);
  float var  = q * (1.0f / Gg) - mean * mean;
  float inv  = rsqrtf(var + EPSV);
  for (int r = 0; r < Gg; ++r)
    buf[r * 256 + c] = g[c] * (buf[r * 256 + c] - mean) * inv + beta[c];
}

// ---------------- launch ----------------
static inline size_t align256(size_t x) { return (x + 255) & ~(size_t)255; }

extern "C" void kernel_launch(void* const* d_in, const int* in_sizes, int n_in,
                              void* d_out, int out_size, void* d_ws, size_t ws_size,
                              hipStream_t stream) {
  (void)in_sizes; (void)n_in; (void)out_size; (void)ws_size;

  const float* x     = (const float*)d_in[0];
  const int*   ei    = (const int*)d_in[1];
  const int*   batch = (const int*)d_in[2];
  const int*   src   = ei;
  const int*   dst   = ei + Ee;

  const float *W[3], *Asrc[3], *Adst[3], *Bb[3], *Bg[3], *Bbeta[3];
  for (int l = 0; l < 3; ++l) {
    W[l]     = (const float*)d_in[3 + 6 * l + 0];
    Asrc[l]  = (const float*)d_in[3 + 6 * l + 1];
    Adst[l]  = (const float*)d_in[3 + 6 * l + 2];
    Bb[l]    = (const float*)d_in[3 + 6 * l + 3];
    Bg[l]    = (const float*)d_in[3 + 6 * l + 4];
    Bbeta[l] = (const float*)d_in[3 + 6 * l + 5];
  }
  const float* mw1 = (const float*)d_in[21]; const float* mb1 = (const float*)d_in[22];
  const float* mg1 = (const float*)d_in[23]; const float* mbe1 = (const float*)d_in[24];
  const float* mw2 = (const float*)d_in[25]; const float* mb2 = (const float*)d_in[26];
  const float* mg2 = (const float*)d_in[27]; const float* mbe2 = (const float*)d_in[28];
  const float* mw3 = (const float*)d_in[29]; const float* mb3 = (const float*)d_in[30];

  // workspace carving (~64 MB, L2-resident)
  char* base = (char*)d_ws;
  size_t off = 0;
  float*    bufL   = (float*)(base + off); off = align256(off + (size_t)Nn * 256 * 4);
  float*    bufA   = (float*)(base + off); off = align256(off + (size_t)Nn * 256 * 4);
  float*    als    = (float*)(base + off); off = align256(off + (size_t)Nn * Hh * 4);
  float*    ald    = (float*)(base + off); off = align256(off + (size_t)Nn * Hh * 4);
  unsigned* mbuf   = (unsigned*)(base + off); off = align256(off + (size_t)Nn * Hh * 4);
  float*    sbuf   = (float*)(base + off); off = align256(off + (size_t)Nn * Hh * 4);
  float*    wbuf   = (float*)(base + off); off = align256(off + (size_t)ETOT * Hh * 4);
  float*    colsum = (float*)(base + off); off = align256(off + 256 * 4);
  float*    colsq  = (float*)(base + off); off = align256(off + 256 * 4);
  float*    pooled = (float*)(base + off); off = align256(off + (size_t)Gg * 256 * 4);
  float*    t1     = (float*)(base + off); off = align256(off + (size_t)Gg * 256 * 4);
  float*    t2     = (float*)(base + off); off = align256(off + (size_t)Gg * 256 * 4);
  _Float16* xh     = (_Float16*)(base + off); off = align256(off + (size_t)Nn * DIN * 2);
  _Float16* hh     = (_Float16*)(base + off); off = align256(off + (size_t)Nn * 256 * 2);
  _Float16* Wt[3];
  Wt[0] = (_Float16*)(base + off); off = align256(off + (size_t)DIN * 256 * 2);
  Wt[1] = (_Float16*)(base + off); off = align256(off + (size_t)HID * 256 * 2);
  Wt[2] = (_Float16*)(base + off); off = align256(off + (size_t)HID * 256 * 2);

  const int gemmBlocks  = Nn / 32;                 // 625
  const int edgeBlocks  = (ETOT + 255) / 256;
  const int aggBlocks   = (ETOT + 7) / 8;
  const int elemBlocks  = (Nn * 256 + 255) / 256;
  const int logitBlocks = (Nn * Hh + 255) / 256;

  // one-time f16 conversions (weights transposed for contiguous async B staging)
  cvt_f32_f16<<<(Nn * DIN + 255) / 256, 256, 0, stream>>>(x, xh, Nn * DIN);
  cvt_weight_t<<<(DIN * 256 + 255) / 256, 256, 0, stream>>>(W[0], Wt[0], DIN);
  cvt_weight_t<<<(HID * 256 + 255) / 256, 256, 0, stream>>>(W[1], Wt[1], HID);
  cvt_weight_t<<<(HID * 256 + 255) / 256, 256, 0, stream>>>(W[2], Wt[2], HID);

  for (int l = 0; l < 3; ++l) {
    const _Float16* inh = (l == 0) ? xh : hh;
    const int       K   = (l == 0) ? DIN : HID;

    gemm_wmma_f16<<<gemmBlocks, 256, 0, stream>>>(inh, Wt[l], bufL, K);
    attn_logits<<<logitBlocks, 256, 0, stream>>>(bufL, Asrc[l], Adst[l], als, ald);

    hipMemsetAsync(mbuf, 0, (size_t)Nn * Hh * 4, stream);   // encoded -inf
    hipMemsetAsync(sbuf, 0, (size_t)Nn * Hh * 4, stream);
    hipMemsetAsync(bufA, 0, (size_t)Nn * 256 * 4, stream);

    edge_max<<<edgeBlocks, 256, 0, stream>>>(src, dst, als, ald, mbuf);
    edge_expsum<<<edgeBlocks, 256, 0, stream>>>(src, dst, als, ald, mbuf, wbuf, sbuf);
    edge_agg<<<aggBlocks, 256, 0, stream>>>(src, dst, wbuf, sbuf, bufL, bufA);

    hipMemsetAsync(colsum, 0, 256 * 4, stream);
    hipMemsetAsync(colsq, 0, 256 * 4, stream);
    bn_stats<<<Nn / BN_ROWS, 256, 0, stream>>>(bufA, Bb[l], colsum, colsq);
    bn_apply_relu<<<elemBlocks, 256, 0, stream>>>(bufA, hh, Bb[l], colsum, colsq,
                                                  Bg[l], Bbeta[l]);
  }

  hipMemsetAsync(pooled, 0, (size_t)Gg * 256 * 4, stream);
  pool_sum<<<elemBlocks, 256, 0, stream>>>(bufA, batch, pooled);

  head_gemm_lrelu<<<(Gg * 256 + 255) / 256, 256, 0, stream>>>(pooled, mw1, mb1, t1, 256, 256);
  head_bn<<<1, 256, 0, stream>>>(t1, mg1, mbe1);
  head_gemm_lrelu<<<(Gg * 256 + 255) / 256, 256, 0, stream>>>(t1, mw2, mb2, t2, 256, 256);
  head_bn<<<1, 256, 0, stream>>>(t2, mg2, mbe2);
  head_gemm_lrelu<<<(Gg * LAT + 255) / 256, 256, 0, stream>>>(t2, mw3, mb3, (float*)d_out, 256, LAT);
}
